// TreeTransformer_2233382994143
// MI455X (gfx1250) — matrix operand order repaired and checked by reading, not compiled, not run on hardware
//
#include <hip/hip_runtime.h>

// ---------------- problem constants ----------------
#define B_    128
#define A_    4
#define N_    128
#define F_IN_ 16
#define H_    256
#define HEADS_ 4
#define FF_   512
#define OUT_  64
#define BF_   3
#define L_    2
#define P_    (N_ * BF_)      // 384
#define T_    (B_ * A_)       // 512
#define E_    (N_ - 1)        // 127
#define DH_   (H_ / HEADS_)   // 64
#define MROWS (T_ * N_)       // 65536

typedef _Float16 h16;
typedef __attribute__((ext_vector_type(16))) _Float16 v16h;
typedef __attribute__((ext_vector_type(8)))  float    v8f;

// ---------------- WMMA fragment loaders (wave32, v_wmma_f32_16x16x32_f16) ----------------
// A (16x32, row-major source, stride in halfs):
//   lane 0-15: M=lane,    halfs[0..15]=K k0..k0+15
//   lane16-31: M=lane-16, halfs[0..15]=K k0+16..k0+31
__device__ __forceinline__ v16h load_frag_a(const h16* __restrict__ base, int stride,
                                            int row0, int k0) {
  int lane = threadIdx.x & 31;
  const h16* p = base + (size_t)(row0 + (lane & 15)) * stride + k0 + ((lane >> 4) * 16);
  union { v16h v; uint4 q[2]; } u;
  u.q[0] = *reinterpret_cast<const uint4*>(p);
  u.q[1] = *reinterpret_cast<const uint4*>(p + 8);
  return u.v;
}
// B (32x16) for D = A x B with B[k][n] = W[n][k], W row-major N x K:
//   lane 0-15: col=lane,    halfs[0..15] = K k0..k0+15
//   lane16-31: col=lane-16, halfs[0..15] = K k0+16..k0+31
__device__ __forceinline__ v16h load_frag_b(const h16* __restrict__ base, int stride,
                                            int col0, int k0) {
  int lane = threadIdx.x & 31;
  const h16* p = base + (size_t)(col0 + (lane & 15)) * stride + k0 + ((lane >> 4) * 16);
  union { v16h v; uint4 q[2]; } u;
  u.q[0] = *reinterpret_cast<const uint4*>(p);
  u.q[1] = *reinterpret_cast<const uint4*>(p + 8);
  return u.v;
}

#define WMMA_F16(a, b, c) \
  __builtin_amdgcn_wmma_f32_16x16x32_f16(false, (a), false, (b), (short)0, (c), false, false)

// ---------------- f32 -> f16 convert ----------------
__global__ void cvt_f16_kernel(const float* __restrict__ s, h16* __restrict__ d, int n) {
  int i = blockIdx.x * 256 + threadIdx.x;
  if (i < n) d[i] = (h16)s[i];
}

// ---------------- input embedding + tree positional encoding ----------------
// one block per tree; 256 threads = H channels.
// pos[n] is a 0/1 vector with one bit per ancestor edge -> gather W_pos columns.
__global__ __launch_bounds__(256) void input_embed_kernel(
    const float* __restrict__ forest, const int* __restrict__ adjacency,
    const int* __restrict__ node_order,
    const float* __restrict__ W_in, const float* __restrict__ b_in,
    const float* __restrict__ W_pos, const float* __restrict__ b_pos,
    float* __restrict__ xf, h16* __restrict__ xh) {
  __shared__ int parentL[N_];
  __shared__ int slotL[N_];
  __shared__ int depthL[N_];
  __shared__ int maxno;
  const int t = blockIdx.x;
  const int tid = threadIdx.x;
  if (tid == 0) { parentL[0] = -1; slotL[0] = 0; maxno = (int)0x80000000; }
  __syncthreads();
  if (tid < E_) {
    const int* e = adjacency + ((size_t)t * E_ + tid) * 3;
    int p = e[0] - t * N_;
    int c = e[1] - t * N_;
    int s = e[2] + 1;
    s = s < 0 ? 0 : (s > BF_ - 1 ? BF_ - 1 : s);
    if (c >= 0 && c < N_) {
      parentL[c] = (p >= 0 && p < N_) ? p : 0;
      slotL[c] = s;
    }
  }
  if (tid < N_) atomicMax(&maxno, node_order[(size_t)t * N_ + tid]);
  __syncthreads();
  if (tid < N_) depthL[tid] = maxno - node_order[(size_t)t * N_ + tid];
  __syncthreads();

  const int h = tid;
  float win[F_IN_];
#pragma unroll
  for (int i = 0; i < F_IN_; ++i) win[i] = W_in[h * F_IN_ + i];
  const float bb = b_in[h] + b_pos[h];
  const float* wpr = W_pos + (size_t)h * P_;

  for (int n = 0; n < N_; ++n) {
    float acc = 0.f;
    int c = n;
    for (int it = 0; it < N_ && c > 0; ++it) {
      int p = parentL[c];
      acc += wpr[depthL[p] * BF_ + slotL[c]];
      c = p;
    }
    const float* fr = forest + ((size_t)t * N_ + n) * F_IN_;
    float x = acc + bb;
#pragma unroll
    for (int i = 0; i < F_IN_; ++i) x += fr[i] * win[i];
    size_t o = ((size_t)t * N_ + n) * H_ + h;
    xf[o] = x;
    xh[o] = (h16)x;
  }
}

// ---------------- generic WMMA GEMM: out[M,N] = A[M,K] @ W[N,K]^T (+bias) ----------------
// MODE 0: outH = acc+bias (f16)        MODE 1: outH = relu(acc+bias) (f16)
// MODE 2: outF += acc+bias (residual)  MODE 3: outF = acc+bias
// block = 256 threads = 8 waves stacked in M; wave tile 32x64 (2x4 WMMA tiles);
// block tile 256x64.  Requires M%256==0, N%64==0, K%32==0 (true for all launches).
template <int MODE>
__global__ __launch_bounds__(256) void gemm_wmma_kernel(
    const h16* __restrict__ Am, const h16* __restrict__ Wm,
    const float* __restrict__ bias,
    h16* __restrict__ outH, float* __restrict__ outF,
    int M, int Nn, int K) {
  const int wv = threadIdx.x >> 5;
  const int lane = threadIdx.x & 31;
  const int rb = blockIdx.y * 256 + wv * 32;
  const int cb = blockIdx.x * 64;
  v8f acc[2][4] = {};
  for (int k = 0; k < K; k += 32) {
    v16h a0 = load_frag_a(Am, K, rb,      k);
    v16h a1 = load_frag_a(Am, K, rb + 16, k);
    v16h b0 = load_frag_b(Wm, K, cb,      k);
    v16h b1 = load_frag_b(Wm, K, cb + 16, k);
    v16h b2 = load_frag_b(Wm, K, cb + 32, k);
    v16h b3 = load_frag_b(Wm, K, cb + 48, k);
    acc[0][0] = WMMA_F16(a0, b0, acc[0][0]);
    acc[0][1] = WMMA_F16(a0, b1, acc[0][1]);
    acc[0][2] = WMMA_F16(a0, b2, acc[0][2]);
    acc[0][3] = WMMA_F16(a0, b3, acc[0][3]);
    acc[1][0] = WMMA_F16(a1, b0, acc[1][0]);
    acc[1][1] = WMMA_F16(a1, b1, acc[1][1]);
    acc[1][2] = WMMA_F16(a1, b2, acc[1][2]);
    acc[1][3] = WMMA_F16(a1, b3, acc[1][3]);
  }
  const int cl = lane & 15;
  const int rofs = (lane >> 4) * 8;
#pragma unroll
  for (int mi = 0; mi < 2; ++mi) {
#pragma unroll
    for (int ni = 0; ni < 4; ++ni) {
      int col = cb + ni * 16 + cl;
      float bv = bias ? bias[col] : 0.f;
#pragma unroll
      for (int j = 0; j < 8; ++j) {
        int row = rb + mi * 16 + rofs + j;
        float v = acc[mi][ni][j] + bv;
        if (MODE == 1) v = v > 0.f ? v : 0.f;
        size_t idx = (size_t)row * Nn + col;
        if (MODE == 2)      outF[idx] += v;
        else if (MODE == 3) outF[idx] = v;
        else                outH[idx] = (h16)v;
      }
    }
  }
}

// ---------------- fused attention: one block per (tree, head) ----------------
__global__ __launch_bounds__(256) void attention_kernel(const h16* __restrict__ qkv,
                                                        h16* __restrict__ att) {
  extern __shared__ char smem[];
  h16* Qs = (h16*)smem;              // 128 x 64
  h16* Ks = Qs + N_ * DH_;           // 128 x 64
  h16* Vt = Ks + N_ * DH_;           // 64 x 128 (V transposed)
  h16* Ps = Vt + DH_ * N_;           // 128 x 128
  const int t  = blockIdx.x >> 2;
  const int hh = blockIdx.x & 3;
  const int tid = threadIdx.x;
  const h16* base = qkv + (size_t)t * N_ * (3 * H_);
#pragma unroll
  for (int i = 0; i < 4; ++i) {
    int u = tid + i * 256;           // 0..1023 (128 rows x 8 chunks)
    int row = u >> 3, ch = (u & 7) * 8;
    *reinterpret_cast<uint4*>(Qs + row * DH_ + ch) =
        *reinterpret_cast<const uint4*>(base + (size_t)row * (3 * H_) + hh * DH_ + ch);
    *reinterpret_cast<uint4*>(Ks + row * DH_ + ch) =
        *reinterpret_cast<const uint4*>(base + (size_t)row * (3 * H_) + H_ + hh * DH_ + ch);
  }
#pragma unroll
  for (int i = 0; i < 32; ++i) {
    int u = tid + i * 256;           // 0..8191
    int row = u >> 6, d = u & 63;
    Vt[d * N_ + row] = base[(size_t)row * (3 * H_) + 2 * H_ + hh * DH_ + d];
  }
  __syncthreads();

  const int wv = tid >> 5;
  const int lane = tid & 31;
  const int cl = lane & 15;
  const int rofs = (lane >> 4) * 8;

  // S = (Q K^T) * (1/sqrt(64)) -> Ps f16
  {
    int rb = wv * 16;
    for (int nt = 0; nt < 8; ++nt) {
      v8f c = {};
#pragma unroll
      for (int ks = 0; ks < 2; ++ks) {
        v16h a = load_frag_a(Qs, DH_, rb, ks * 32);
        v16h b = load_frag_b(Ks, DH_, nt * 16, ks * 32);
        c = WMMA_F16(a, b, c);
      }
      int col = nt * 16 + cl;
#pragma unroll
      for (int j = 0; j < 8; ++j)
        Ps[(rb + rofs + j) * N_ + col] = (h16)(c[j] * 0.125f);
    }
  }
  __syncthreads();
  if (tid < N_) {
    h16* pr = Ps + tid * N_;
    float mx = -1e30f;
    for (int c2 = 0; c2 < N_; ++c2) { float v = (float)pr[c2]; mx = v > mx ? v : mx; }
    float s = 0.f;
    for (int c2 = 0; c2 < N_; ++c2) { float e = __expf((float)pr[c2] - mx); s += e; pr[c2] = (h16)e; }
    float inv = 1.f / s;
    for (int c2 = 0; c2 < N_; ++c2) pr[c2] = (h16)((float)pr[c2] * inv);
  }
  __syncthreads();
  // O = P @ V
  {
    int rb = wv * 16;
    for (int nt = 0; nt < 4; ++nt) {
      v8f c = {};
#pragma unroll
      for (int ks = 0; ks < 4; ++ks) {
        v16h a = load_frag_a(Ps, N_, rb, ks * 32);
        v16h b = load_frag_b(Vt, N_, nt * 16, ks * 32);
        c = WMMA_F16(a, b, c);
      }
      int col = nt * 16 + cl;
#pragma unroll
      for (int j = 0; j < 8; ++j) {
        int row = rb + rofs + j;
        att[((size_t)t * N_ + row) * H_ + hh * DH_ + col] = (h16)c[j];
      }
    }
  }
}

// ---------------- wave-per-row LayerNorm over H=256 (wave32 shuffles) ----------------
__global__ __launch_bounds__(256) void layernorm_kernel(float* __restrict__ xf,
                                                        h16* __restrict__ xh,
                                                        const float* __restrict__ w,
                                                        const float* __restrict__ b) {
  const int wv = threadIdx.x >> 5;
  const int lane = threadIdx.x & 31;
  const int row = blockIdx.x * 8 + wv;
  float* xr = xf + (size_t)row * H_;
  float vals[8];
  float s = 0.f;
#pragma unroll
  for (int j = 0; j < 8; ++j) { vals[j] = xr[lane + j * 32]; s += vals[j]; }
#pragma unroll
  for (int m = 16; m > 0; m >>= 1) s += __shfl_xor(s, m, 32);
  const float mu = s * (1.f / H_);
  float vs = 0.f;
#pragma unroll
  for (int j = 0; j < 8; ++j) { float d = vals[j] - mu; vs += d * d; }
#pragma unroll
  for (int m = 16; m > 0; m >>= 1) vs += __shfl_xor(vs, m, 32);
  const float inv = rsqrtf(vs * (1.f / H_) + 1e-5f);
#pragma unroll
  for (int j = 0; j < 8; ++j) {
    int i = lane + j * 32;
    float y = (vals[j] - mu) * inv * w[i] + b[i];
    xr[i] = y;
    xh[(size_t)row * H_ + i] = (h16)y;
  }
}

// ---------------- final layernorm over OUT=64 ----------------
__global__ void final_ln_kernel(const float* __restrict__ ob,
                                const float* __restrict__ w, const float* __restrict__ b,
                                float* __restrict__ dout) {
  __shared__ float smu, sinv;
  const int r = blockIdx.x, i = threadIdx.x;
  if (i == 0) {
    float s = 0.f;
    for (int j = 0; j < OUT_; ++j) s += ob[(size_t)r * OUT_ + j];
    float m = s * (1.f / OUT_);
    float v = 0.f;
    for (int j = 0; j < OUT_; ++j) { float d = ob[(size_t)r * OUT_ + j] - m; v += d * d; }
    smu = m;
    sinv = rsqrtf(v * (1.f / OUT_) + 1e-5f);
  }
  __syncthreads();
  dout[(size_t)r * OUT_ + i] = (ob[(size_t)r * OUT_ + i] - smu) * sinv * w[i] + b[i];
}

// ---------------- host launcher ----------------
extern "C" void kernel_launch(void* const* d_in, const int* in_sizes, int n_in,
                              void* d_out, int out_size, void* d_ws, size_t ws_size,
                              hipStream_t stream) {
  const float* forest     = (const float*)d_in[0];
  const int*   adjacency  = (const int*)  d_in[1];
  const int*   node_order = (const int*)  d_in[2];
  const float* W_in   = (const float*)d_in[3];
  const float* b_in   = (const float*)d_in[4];
  const float* W_pos  = (const float*)d_in[5];
  const float* b_pos  = (const float*)d_in[6];
  const float* qkv_w  = (const float*)d_in[7];
  const float* qkv_b  = (const float*)d_in[8];
  const float* outp_w = (const float*)d_in[9];
  const float* outp_b = (const float*)d_in[10];
  const float* ln1_w  = (const float*)d_in[11];
  const float* ln1_b  = (const float*)d_in[12];
  const float* ff1_w  = (const float*)d_in[13];
  const float* ff1_b  = (const float*)d_in[14];
  const float* ff2_w  = (const float*)d_in[15];
  const float* ff2_b  = (const float*)d_in[16];
  const float* ln2_w  = (const float*)d_in[17];
  const float* ln2_b  = (const float*)d_in[18];
  const float* W_out  = (const float*)d_in[19];
  const float* b_out  = (const float*)d_in[20];
  const float* lnf_w  = (const float*)d_in[21];
  const float* lnf_b  = (const float*)d_in[22];

  char* ws = (char*)d_ws;
  size_t off = 0;
  auto alloc = [&](size_t bytes) -> void* {
    void* p = ws + off;
    off = (off + bytes + 255) & ~(size_t)255;
    return p;
  };
  float* xf    = (float*)alloc((size_t)MROWS * H_ * 4);
  h16*   xh    = (h16*)  alloc((size_t)MROWS * H_ * 2);
  h16*   qkvh  = (h16*)  alloc((size_t)MROWS * 3 * H_ * 2);   // reused as FF hidden
  h16*   atth  = (h16*)  alloc((size_t)MROWS * H_ * 2);
  float* outb  = (float*)alloc((size_t)T_ * OUT_ * 4);
  h16*   wqkvh = (h16*)  alloc((size_t)L_ * 3 * H_ * H_ * 2);
  h16*   wouth = (h16*)  alloc((size_t)L_ * H_ * H_ * 2);
  h16*   wff1h = (h16*)  alloc((size_t)L_ * FF_ * H_ * 2);
  h16*   wff2h = (h16*)  alloc((size_t)L_ * H_ * FF_ * 2);
  h16*   wfinh = (h16*)  alloc((size_t)OUT_ * N_ * H_ * 2);
  h16*   ffh   = qkvh;

  auto cvt = [&](const float* s, h16* d, int n) {
    cvt_f16_kernel<<<(n + 255) / 256, 256, 0, stream>>>(s, d, n);
  };
  cvt(qkv_w,  wqkvh, L_ * 3 * H_ * H_);
  cvt(outp_w, wouth, L_ * H_ * H_);
  cvt(ff1_w,  wff1h, L_ * FF_ * H_);
  cvt(ff2_w,  wff2h, L_ * H_ * FF_);
  cvt(W_out,  wfinh, OUT_ * N_ * H_);

  input_embed_kernel<<<T_, 256, 0, stream>>>(forest, adjacency, node_order,
                                             W_in, b_in, W_pos, b_pos, xf, xh);

  const size_t att_smem = (size_t)(3 * N_ * DH_ + N_ * N_) * sizeof(h16); // 80 KB

  for (int l = 0; l < L_; ++l) {
    gemm_wmma_kernel<0><<<dim3((3 * H_) / 64, MROWS / 256), 256, 0, stream>>>(
        xh, wqkvh + (size_t)l * 3 * H_ * H_, qkv_b + (size_t)l * 3 * H_,
        qkvh, nullptr, MROWS, 3 * H_, H_);
    attention_kernel<<<T_ * HEADS_, 256, att_smem, stream>>>(qkvh, atth);
    gemm_wmma_kernel<2><<<dim3(H_ / 64, MROWS / 256), 256, 0, stream>>>(
        atth, wouth + (size_t)l * H_ * H_, outp_b + (size_t)l * H_,
        nullptr, xf, MROWS, H_, H_);
    layernorm_kernel<<<MROWS / 8, 256, 0, stream>>>(xf, xh,
        ln1_w + (size_t)l * H_, ln1_b + (size_t)l * H_);
    gemm_wmma_kernel<1><<<dim3(FF_ / 64, MROWS / 256), 256, 0, stream>>>(
        xh, wff1h + (size_t)l * FF_ * H_, ff1_b + (size_t)l * FF_,
        ffh, nullptr, MROWS, FF_, H_);
    gemm_wmma_kernel<2><<<dim3(H_ / 64, MROWS / 256), 256, 0, stream>>>(
        ffh, wff2h + (size_t)l * H_ * FF_, ff2_b + (size_t)l * H_,
        nullptr, xf, MROWS, H_, FF_);
    layernorm_kernel<<<MROWS / 8, 256, 0, stream>>>(xf, xh,
        ln2_w + (size_t)l * H_, ln2_b + (size_t)l * H_);
  }

  // readout: xh viewed as (512 x 32768) @ W_out^T (32768 x 64) + b_out
  gemm_wmma_kernel<3><<<dim3(OUT_ / 64, T_ / 256), 256, 0, stream>>>(
      xh, wfinh, b_out, nullptr, outb, T_, OUT_, N_ * H_);
  final_ln_kernel<<<T_, OUT_, 0, stream>>>(outb, lnf_w, lnf_b, (float*)d_out);
}
